// Factum_81595788689998
// MI455X (gfx1250) — matrix operand order repaired and checked by reading, not compiled, not run on hardware
//
#include <hip/hip_runtime.h>
#include <hip/hip_bf16.h>

// ---------------------------------------------------------------------------
// Problem constants (from reference): B=8, S=2048, D=1024, H=512, C=256, E=64
// ---------------------------------------------------------------------------
constexpr int BB = 8;
constexpr int SS = 2048;
constexpr int DD = 1024;
constexpr int HH = 512;
constexpr int CC = 256;
constexpr int EE = 64;
constexpr int NN = BB * SS;          // 16384 nodes per side

typedef __attribute__((ext_vector_type(16))) _Float16 v16h;
typedef __attribute__((ext_vector_type(8)))  _Float16 v8h;
typedef __attribute__((ext_vector_type(8)))  float    v8f;

union Frag16 { v16h v; v8h h[2]; };

// ---------------------------------------------------------------------------
// Weight prep: cast f32 [K,N] -> f16 [N,K] (transposed, row-major-over-N)
// ---------------------------------------------------------------------------
__global__ __launch_bounds__(256) void cast_transpose_kernel(
    const float* __restrict__ in, _Float16* __restrict__ out, int K, int N) {
  int idx = blockIdx.x * 256 + threadIdx.x;
  if (idx >= K * N) return;
  int k = idx / N;
  int n = idx - k * N;
  out[(size_t)n * K + k] = (_Float16)in[idx];
}

__global__ __launch_bounds__(256) void concat_bias_kernel(
    const float* __restrict__ a, const float* __restrict__ b, float* __restrict__ out) {
  int i = blockIdx.x * 256 + threadIdx.x;   // 512 total
  out[i] = (i < CC) ? a[i] : b[i - CC];
}

// Etab[v][c] = sum_e emb[v][e] * W_edge[e][c]   (64 x 256, K=64)
__global__ __launch_bounds__(256) void etab_kernel(
    const float* __restrict__ emb, const float* __restrict__ wedge,
    float* __restrict__ etab) {
  int v = blockIdx.x;       // 0..63
  int c = threadIdx.x;      // 0..255
  float s = 0.f;
#pragma unroll 8
  for (int e = 0; e < EE; ++e) s += emb[v * EE + e] * wedge[e * CC + c];
  etab[v * CC + c] = s;
}

__global__ __launch_bounds__(256) void zero_kernel(float* __restrict__ p, int n) {
  int i = blockIdx.x * 256 + threadIdx.x;
  if (i < n) p[i] = 0.f;
}

// ---------------------------------------------------------------------------
// Tiled WMMA GEMM:  C[M,N] = act(A[M,K] @ Wt^T + bias)
//   A: f32 or f16, row-major [M,K]
//   Wt: f16 [N,K]  (pre-transposed weights)
//   block tile 128x128, 8 waves (2 frags M x 4 frags N each), KSTEP=32
// ---------------------------------------------------------------------------
constexpr int LDS_STRIDE = 40;   // 32 + 8 pad halfs -> conflict-free b128 reads

template <bool A_IS_F16, bool RELU, bool OUT_F16>
__global__ __launch_bounds__(256) void gemm_wmma_kernel(
    const void* __restrict__ Aptr, const _Float16* __restrict__ Bt,
    const float* __restrict__ bias, void* __restrict__ Cptr,
    int M, int N, int K) {
  __shared__ _Float16 As[128 * LDS_STRIDE];
  __shared__ _Float16 Bs[128 * LDS_STRIDE];

  const int tid  = threadIdx.x;
  const int m0   = blockIdx.y * 128;
  const int n0   = blockIdx.x * 128;
  const int wave = tid >> 5;
  const int lane = tid & 31;
  const int wm   = wave & 3;          // M sub-tile 0..3  (x32)
  const int wn   = wave >> 2;         // N sub-tile 0..1  (x64)
  const int l16  = lane & 15;
  const bool hi  = lane >= 16;

  // tile-load task: each thread owns 16 contiguous K elements of one row
  const int lr = tid >> 1;            // row within tile 0..127
  const int lh = tid & 1;             // which 16-element half of the 32-wide K slab

  v8f acc[2][4] = {};

  for (int k0 = 0; k0 < K; k0 += 32) {
    // ---- stage A tile (convert f32->f16 if needed) ----
    if constexpr (A_IS_F16) {
      const _Float16* A = (const _Float16*)Aptr;
      const v8h* src = (const v8h*)(A + (size_t)(m0 + lr) * K + k0 + lh * 16);
      _Float16* dst = &As[lr * LDS_STRIDE + lh * 16];
      *(v8h*)dst = src[0];
      *(v8h*)(dst + 8) = src[1];
    } else {
      const float* A = (const float*)Aptr;
      const float4* src = (const float4*)(A + (size_t)(m0 + lr) * K + k0 + lh * 16);
      float tmp[16];
#pragma unroll
      for (int j = 0; j < 4; ++j) {
        float4 f = src[j];
        tmp[4 * j + 0] = f.x; tmp[4 * j + 1] = f.y;
        tmp[4 * j + 2] = f.z; tmp[4 * j + 3] = f.w;
      }
      _Float16* dst = &As[lr * LDS_STRIDE + lh * 16];
#pragma unroll
      for (int j = 0; j < 16; ++j) dst[j] = (_Float16)tmp[j];
      if (k0 + 32 < K) __builtin_prefetch((const float*)src + 32, 0, 0);
    }
    // ---- stage B tile (Wt is already f16 [N,K]) ----
    {
      const v8h* src = (const v8h*)(Bt + (size_t)(n0 + lr) * K + k0 + lh * 16);
      _Float16* dst = &Bs[lr * LDS_STRIDE + lh * 16];
      *(v8h*)dst = src[0];
      *(v8h*)(dst + 8) = src[1];
    }
    __syncthreads();

    // ---- fragment loads per ISA 7.12.2 layouts ----
    Frag16 af[2];
#pragma unroll
    for (int i = 0; i < 2; ++i) {
      const _Float16* p =
          &As[(wm * 32 + i * 16 + l16) * LDS_STRIDE + (hi ? 8 : 0)];
      af[i].h[0] = *(const v8h*)p;          // K = kb .. kb+7
      af[i].h[1] = *(const v8h*)(p + 16);   // K = kb+16 .. kb+23
    }
    Frag16 bfr[4];
#pragma unroll
    for (int j = 0; j < 4; ++j) {
      const _Float16* p =
          &Bs[(wn * 64 + j * 16 + l16) * LDS_STRIDE + (hi ? 8 : 0)];
      bfr[j].h[0] = *(const v8h*)p;         // K = kb .. kb+7
      bfr[j].h[1] = *(const v8h*)(p + 16);  // K = kb+16 .. kb+23
    }

#pragma unroll
    for (int i = 0; i < 2; ++i)
#pragma unroll
      for (int j = 0; j < 4; ++j)
        acc[i][j] = __builtin_amdgcn_wmma_f32_16x16x32_f16(
            false, af[i].v, false, bfr[j].v, (short)0, acc[i][j], false, false);

    __syncthreads();
  }

  // ---- epilogue: bias + optional leaky_relu, f16 or f32 store ----
#pragma unroll
  for (int i = 0; i < 2; ++i) {
    const int rbase = m0 + wm * 32 + i * 16 + (hi ? 8 : 0);
#pragma unroll
    for (int j = 0; j < 4; ++j) {
      const int gcol = n0 + wn * 64 + j * 16 + l16;
      const float bv = bias[gcol];
#pragma unroll
      for (int v = 0; v < 8; ++v) {
        float val = acc[i][j][v] + bv;
        if constexpr (RELU) val = (val > 0.f) ? val : 0.01f * val;
        const size_t off = (size_t)(rbase + v) * N + gcol;
        if constexpr (OUT_F16) ((_Float16*)Cptr)[off] = (_Float16)val;
        else                   ((float*)Cptr)[off] = val;
      }
    }
  }
}

// ---------------------------------------------------------------------------
// Graph gather + L2-normalize + per-batch mean accumulation.
// XMS[n] = [ x@W_msg+b_msg (256) | x@W_self+b_self (256) ]
// out[n] = self + valid*(msg[head] + Etab[wtype*wmask]); normalize; sum into racc
// One block = 64 nodes of one batch; 8 waves x 8 nodes; lane owns 8 channels.
// ---------------------------------------------------------------------------
__global__ __launch_bounds__(256) void combine_kernel(
    const float* __restrict__ XMS, const float* __restrict__ Etab,
    const int* __restrict__ dense, const int* __restrict__ sparse,
    const int* __restrict__ arc_head, const int* __restrict__ wtype,
    const int* __restrict__ wmask, float* __restrict__ racc) {
  __shared__ float lacc[CC];
  const int tid = threadIdx.x;
  lacc[tid] = 0.f;
  __syncthreads();

  constexpr int BLKS_PER_BATCH = SS / 64;   // 32
  const int b = blockIdx.x / BLKS_PER_BATCH;
  const int chunk = blockIdx.x - b * BLKS_PER_BATCH;
  const int wave = tid >> 5;
  const int lane = tid & 31;

  float acc[8] = {0.f, 0.f, 0.f, 0.f, 0.f, 0.f, 0.f, 0.f};

  for (int t = 0; t < 8; ++t) {
    const int sidx = chunk * 64 + wave * 8 + t;
    const int n = b * SS + sidx;
    const float va = (dense[n] * sparse[n]) ? 1.f : 0.f;
    const int g = b * SS + arc_head[n];
    const int wt = wtype[n] * wmask[n];

    float vals[8];
    float ss = 0.f;
#pragma unroll
    for (int j = 0; j < 8; ++j) {
      const int c = lane + 32 * j;
      float v = XMS[(size_t)n * 512 + 256 + c] +
                va * (XMS[(size_t)g * 512 + c] + Etab[wt * CC + c]);
      vals[j] = v;
      ss += v * v;
    }
#pragma unroll
    for (int o = 16; o > 0; o >>= 1) ss += __shfl_xor(ss, o, 32);
    const float scale = 1.f / fmaxf(sqrtf(ss), 1e-12f);
#pragma unroll
    for (int j = 0; j < 8; ++j) acc[j] += vals[j] * scale;
  }

#pragma unroll
  for (int j = 0; j < 8; ++j) atomicAdd(&lacc[lane + 32 * j], acc[j]);
  __syncthreads();
  atomicAdd(&racc[b * CC + tid], lacc[tid]);
}

// ---------------------------------------------------------------------------
// Final projection: out[side,b,:] = (racc[side,b,:]/S) @ Wf + bf
// blocks 0..7 = src batches, 8..15 = tgt batches; d_out layout matches.
// ---------------------------------------------------------------------------
__global__ __launch_bounds__(256) void final_kernel(
    const float* __restrict__ racc, const float* __restrict__ Wf,
    const float* __restrict__ bf, float* __restrict__ out) {
  __shared__ float r[CC];
  const int sb = blockIdx.x;      // 0..15
  const int tid = threadIdx.x;
  r[tid] = racc[sb * CC + tid] * (1.f / (float)SS);
  __syncthreads();
  float s = bf[tid];
#pragma unroll 4
  for (int c = 0; c < CC; ++c) s += r[c] * Wf[c * CC + tid];
  out[sb * CC + tid] = s;
}

// ---------------------------------------------------------------------------
// Orchestration
// ---------------------------------------------------------------------------
extern "C" void kernel_launch(void* const* d_in, const int* in_sizes, int n_in,
                              void* d_out, int out_size, void* d_ws, size_t ws_size,
                              hipStream_t stream) {
  (void)in_sizes; (void)n_in; (void)out_size; (void)ws_size;

  const float* h[2]      = {(const float*)d_in[0], (const float*)d_in[1]};
  const int* dense[2]    = {(const int*)d_in[4],  (const int*)d_in[11]};
  const int* sparse[2]   = {(const int*)d_in[5],  (const int*)d_in[12]};
  const int* head[2]     = {(const int*)d_in[7],  (const int*)d_in[14]};
  const int* wtype[2]    = {(const int*)d_in[9],  (const int*)d_in[16]};
  const int* wmask[2]    = {(const int*)d_in[10], (const int*)d_in[17]};
  const float* W1[2]     = {(const float*)d_in[18], (const float*)d_in[22]};
  const float* b1[2]     = {(const float*)d_in[19], (const float*)d_in[23]};
  const float* W2[2]     = {(const float*)d_in[20], (const float*)d_in[24]};
  const float* b2[2]     = {(const float*)d_in[21], (const float*)d_in[25]};
  const float* emb       = (const float*)d_in[26];
  const float* W_msg     = (const float*)d_in[27];
  const float* b_msg     = (const float*)d_in[28];
  const float* W_edge    = (const float*)d_in[29];
  const float* W_self    = (const float*)d_in[31];
  const float* b_self    = (const float*)d_in[32];
  const float* Wf        = (const float*)d_in[33];
  const float* bf        = (const float*)d_in[34];

  // --- workspace carve-up ---
  char* ws = (char*)d_ws;
  size_t off = 0;
  auto alloc = [&](size_t bytes) {
    void* p = ws + off;
    off += (bytes + 255) & ~(size_t)255;
    return p;
  };
  _Float16* Wt1[2]; _Float16* Wt2[2];
  Wt1[0] = (_Float16*)alloc((size_t)HH * DD * 2);
  Wt1[1] = (_Float16*)alloc((size_t)HH * DD * 2);
  Wt2[0] = (_Float16*)alloc((size_t)CC * HH * 2);
  Wt2[1] = (_Float16*)alloc((size_t)CC * HH * 2);
  _Float16* WtMS = (_Float16*)alloc((size_t)512 * CC * 2);   // rows 0-255 msg, 256-511 self
  float* biasMS  = (float*)alloc(512 * 4);
  float* Etab    = (float*)alloc((size_t)EE * CC * 4);
  float* racc    = (float*)alloc((size_t)2 * BB * CC * 4);
  _Float16* X1   = (_Float16*)alloc((size_t)NN * HH * 2);    // reused per side
  _Float16* X    = (_Float16*)alloc((size_t)NN * CC * 2);    // reused per side
  float* XMS     = (float*)alloc((size_t)NN * 512 * 4);      // reused per side

  // --- one-time prep ---
  for (int s = 0; s < 2; ++s) {
    cast_transpose_kernel<<<(DD * HH + 255) / 256, 256, 0, stream>>>(W1[s], Wt1[s], DD, HH);
    cast_transpose_kernel<<<(HH * CC + 255) / 256, 256, 0, stream>>>(W2[s], Wt2[s], HH, CC);
  }
  cast_transpose_kernel<<<(CC * CC + 255) / 256, 256, 0, stream>>>(W_msg, WtMS, CC, CC);
  cast_transpose_kernel<<<(CC * CC + 255) / 256, 256, 0, stream>>>(W_self, WtMS + (size_t)CC * CC, CC, CC);
  concat_bias_kernel<<<2, 256, 0, stream>>>(b_msg, b_self, biasMS);
  etab_kernel<<<EE, 256, 0, stream>>>(emb, W_edge, Etab);
  zero_kernel<<<(2 * BB * CC + 255) / 256, 256, 0, stream>>>(racc, 2 * BB * CC);

  // --- per side pipeline (sequential; buffers reused) ---
  for (int s = 0; s < 2; ++s) {
    dim3 blk(256);
    // GEMM1: X1 = leaky_relu(h @ W1 + b1)  [16384,1024]x[1024,512] -> f16
    gemm_wmma_kernel<false, true, true><<<dim3(HH / 128, NN / 128), blk, 0, stream>>>(
        h[s], Wt1[s], b1[s], X1, NN, HH, DD);
    // GEMM2: X = X1 @ W2 + b2  [16384,512]x[512,256] -> f16
    gemm_wmma_kernel<true, false, true><<<dim3(CC / 128, NN / 128), blk, 0, stream>>>(
        X1, Wt2[s], b2[s], X, NN, CC, HH);
    // GEMM3: XMS = X @ [W_msg|W_self] + [b_msg|b_self]  [16384,256]x[256,512] -> f32
    gemm_wmma_kernel<true, false, false><<<dim3(512 / 128, NN / 128), blk, 0, stream>>>(
        X, WtMS, biasMS, XMS, NN, 512, CC);
    // gather + normalize + batch-mean accumulate
    combine_kernel<<<BB * (SS / 64), blk, 0, stream>>>(
        XMS, Etab, dense[s], sparse[s], head[s], wtype[s], wmask[s],
        racc + (size_t)s * BB * CC);
  }

  // --- final projection into d_out (src then tgt, contiguous) ---
  final_kernel<<<16, 256, 0, stream>>>(racc, Wf, bf, (float*)d_out);
}
